// Net_7138235646328
// MI455X (gfx1250) — matrix operand order repaired and checked by reading, not compiled, not run on hardware
//
#include <hip/hip_runtime.h>
#include <hip/hip_bf16.h>
#include <math.h>

typedef unsigned short u16;
typedef unsigned int   u32;
typedef unsigned long long u64;

typedef __attribute__((ext_vector_type(8)))  __bf16 v8bf;
typedef __attribute__((ext_vector_type(16))) __bf16 v16bf;
typedef __attribute__((ext_vector_type(8)))  float  v8f;

#define T_STEPS 2048
#define IO_DIM  64
#define L_DIM   1024
#define G3      3072        // 3*L
#define DEPTH   6
#define NWG     64          // persistent grid: 64 WGs x 16 hidden units = 1024
#define TPB     192         // 6 waves: (gate r/z/n) x (matrix ih/hh)

union B8 { uint4 u; v8bf b; };

__device__ inline v8bf ld8(const u16* p) { B8 t; t.u = *(const uint4*)p; return t.b; }

__device__ inline v16bf cat16(v8bf lo, v8bf hi) {
  return __builtin_shufflevector(lo, hi, 0,1,2,3,4,5,6,7,8,9,10,11,12,13,14,15);
}

__device__ inline u16 f2b(float f) {          // fp32 -> bf16 bits, round-nearest-even
  u32 u = __float_as_uint(f);
  return (u16)((u + 0x7FFFu + ((u >> 16) & 1u)) >> 16);
}

// Async memory->LDS copy, 16 bytes per active lane (CDNA5 GLOBAL_LOAD_ASYNC_TO_LDS).
// Tracked by ASYNCcnt; no VGPR data path. LDS address = low 32 bits of the
// generic pointer (ISA 10.2: generic LDS addr = {aperture, lds_offset[31:0]}).
__device__ inline void async_ld16(u32 lds_off, const void* gptr) {
  asm volatile("global_load_async_to_lds_b128 %0, %1, off"
               :: "v"(lds_off), "v"((u64)(uintptr_t)gptr)
               : "memory");
}
__device__ inline void wait_async0() {
  asm volatile("s_wait_asynccnt 0x0" ::: "memory");
}

// ---------------------------------------------------------------------------
// fp32 -> bf16 bulk conversion (weights converted once per launch; stay in L2)
// ---------------------------------------------------------------------------
__global__ void f32_to_bf16(const float* __restrict__ src, u16* __restrict__ dst, size_t n) {
  size_t i = (size_t)blockIdx.x * blockDim.x + threadIdx.x;
  size_t stride = (size_t)gridDim.x * blockDim.x;
  for (; i < n; i += stride) dst[i] = f2b(src[i]);
}

__global__ void init_state(u16* hA, u16* hB, u32* bar) {
  int i = blockIdx.x * blockDim.x + threadIdx.x;
  if (i < DEPTH * L_DIM) { hA[i] = 0; hB[i] = 0; }
  if (i < 2) bar[i] = 0;
}

// ---------------------------------------------------------------------------
// Generic bf16 WMMA GEMM: C[M,N] = A[M,K] * W[N,K]^T (+bias). One 16x16 tile
// per wave, grid-stride over tiles; two independent accumulator chains to
// pipeline consecutive WMMAs. Used for lin1 (batched over T), the layer-0
// W_ih precompute (time-batched), and the final lin2 over all T.
// ---------------------------------------------------------------------------
__global__ void __launch_bounds__(256)
gemm_bf16(const u16* __restrict__ A, const u16* __restrict__ W,
          const float* __restrict__ bias, float* __restrict__ Cf,
          u16* __restrict__ Cb, int M, int N, int K) {
  int lane  = threadIdx.x & 31;
  int wave  = (blockIdx.x * blockDim.x + threadIdx.x) >> 5;
  int nWav  = (gridDim.x * blockDim.x) >> 5;
  int half  = lane >> 4;              // 0: lanes 0-15, 1: lanes 16-31
  int sub   = lane & 15;
  int akoff = half ? 8 : 0;           // A operand per-lane K pattern
  int bkoff = half ? 8 : 0;           // B operand per-lane K pattern
  int tilesN = N >> 4;
  int nTiles = (M >> 4) * tilesN;

  for (int tile = wave; tile < nTiles; tile += nWav) {
    int m0 = (tile / tilesN) << 4;
    int n0 = (tile % tilesN) << 4;
    const u16* Ar = A + (size_t)(m0 + sub) * K + akoff;   // A rows (M side)
    const u16* Wr = W + (size_t)(n0 + sub) * K + bkoff;   // B columns (N side)
    v8f acc0 = {}, acc1 = {};
    for (int kc = 0; kc < K; kc += 64) {
      v16bf a0 = cat16(ld8(Ar + kc), ld8(Ar + kc + 16));
      v16bf b0 = cat16(ld8(Wr + kc), ld8(Wr + kc + 16));
      v16bf a1 = cat16(ld8(Ar + kc + 32), ld8(Ar + kc + 48));
      v16bf b1 = cat16(ld8(Wr + kc + 32), ld8(Wr + kc + 48));
      acc0 = __builtin_amdgcn_wmma_f32_16x16x32_bf16(false, a0, false, b0,
                                                     (short)0, acc0, false, false);
      acc1 = __builtin_amdgcn_wmma_f32_16x16x32_bf16(false, a1, false, b1,
                                                     (short)0, acc1, false, false);
    }
    int n = n0 + sub;                           // D: lane -> column n0+sub
    float bv = bias ? bias[n] : 0.f;
    for (int j = 0; j < 8; ++j) {               // rows j (+8 for upper half)
      int m = m0 + j + (half ? 8 : 0);
      float v = acc0[j] + acc1[j] + bv;
      if (Cf) Cf[(size_t)m * N + n] = v;
      if (Cb) Cb[(size_t)m * N + n] = f2b(v);
    }
  }
}

// ---------------------------------------------------------------------------
// Grid-wide barrier (release/acquire) for the persistent recurrent kernel.
// ---------------------------------------------------------------------------
__device__ inline void gridBarrier(u32* cnt, u32* gen, u32 nwg) {
  __builtin_amdgcn_fence(__ATOMIC_RELEASE, "agent");
  __syncthreads();
  if (threadIdx.x == 0) {
    u32 g = __hip_atomic_load(gen, __ATOMIC_RELAXED, __HIP_MEMORY_SCOPE_AGENT);
    u32 a = __hip_atomic_fetch_add(cnt, 1u, __ATOMIC_ACQ_REL, __HIP_MEMORY_SCOPE_AGENT) + 1u;
    if (a == nwg) {
      __hip_atomic_store(cnt, 0u, __ATOMIC_RELAXED, __HIP_MEMORY_SCOPE_AGENT);
      __hip_atomic_fetch_add(gen, 1u, __ATOMIC_RELEASE, __HIP_MEMORY_SCOPE_AGENT);
    } else {
      while (__hip_atomic_load(gen, __ATOMIC_ACQUIRE, __HIP_MEMORY_SCOPE_AGENT) == g)
        __builtin_amdgcn_s_sleep(1);
    }
  }
  __syncthreads();
  __builtin_amdgcn_fence(__ATOMIC_ACQUIRE, "agent");
}

// ---------------------------------------------------------------------------
// Persistent recurrent core. 64 WGs, each owns hidden rows [wg*16, wg*16+16)
// of every layer. 6 waves/WG: wave w -> gate g=w%3, matrix mat=w/3 (0=ih,1=hh).
// Each wave does a 16x1024 WMMA GEMV: weight tile = A operand, vector chunk
// replicated into all 16 B columns. Vectors are staged into LDS with CDNA5
// async loads (ASYNCcnt). h is double-buffered per step (ping/pong) so one
// grid barrier per (t,layer) suffices. Layer-0 ih-gates come from the
// time-batched GI0 precompute.
// ---------------------------------------------------------------------------
__global__ void __launch_bounds__(TPB)
gru_persistent(const u16* __restrict__ wIh, const u16* __restrict__ wHh,
               const float* __restrict__ bih, const float* __restrict__ bhh,
               const float* __restrict__ GI0,
               u16* __restrict__ hA, u16* __restrict__ hB,
               u16* __restrict__ H5, u32* barCnt, u32* barGen) {
  __shared__ __align__(16) u16 vecA[L_DIM];   // input vector a = h[l-1] (this t)
  __shared__ __align__(16) u16 vecH[L_DIM];   // recurrent vector h[l] (t-1)
  __shared__ float gi[3][16], gh[3][16];
  __shared__ float hold[DEPTH][16];           // this WG's fp32 h slice

  const int tid  = threadIdx.x;
  const int wg   = blockIdx.x;
  const int w    = tid >> 5;
  const int lane = tid & 31;
  const int g    = w % 3;
  const int mat  = w / 3;
  const int half = lane >> 4;
  const int sub  = lane & 15;
  const int akoff = half ? 8 : 0;
  const int bkoff = half ? 8 : 0;
  const int row0 = g * L_DIM + wg * 16;       // this wave's gate-row block
  const u32 ldsH = (u32)(uintptr_t)vecH;      // LDS byte offsets for async DMA
  const u32 ldsA = (u32)(uintptr_t)vecA;

  if (tid < DEPTH * 16) hold[tid >> 4][tid & 15] = 0.f;
  __syncthreads();

  for (int t = 0; t < T_STEPS; ++t) {
    u16* hOld = (t & 1) ? hB : hA;
    u16* hNew = (t & 1) ? hA : hB;
    for (int l = 0; l < DEPTH; ++l) {
      // Stage vectors into LDS with async memory->LDS DMA (no VGPR round-trip).
      // hOld[l] is read-only this step; hNew[l-1] completed before the
      // previous grid barrier.
      {
        const char* srcH = (const char*)(hOld + l * L_DIM);
        const char* srcA = (const char*)(hNew + (l - 1) * L_DIM);
        int nCh = (l > 0) ? 256 : 128;        // 16B chunks: 128 per vector
        for (int i = tid; i < nCh; i += TPB) {
          if (i < 128) async_ld16(ldsH + i * 16, srcH + i * 16);
          else         async_ld16(ldsA + (i - 128) * 16, srcA + (i - 128) * 16);
        }
        wait_async0();
      }
      __syncthreads();

      if (mat == 0 && l == 0) {
        // layer-0 ih gates precomputed (time-batched GEMM)
        if (lane < 16)
          gi[g][lane] = GI0[(size_t)t * G3 + g * L_DIM + wg * 16 + lane];
      } else {
        const u16* Wb = (mat == 0 ? wIh : wHh)
                        + (size_t)l * G3 * L_DIM + (size_t)(row0 + sub) * L_DIM + akoff;
        const u16* vec = ((mat == 0) ? vecA : vecH) + bkoff;
        v8f acc0 = {}, acc1 = {};
        for (int kc = 0; kc < L_DIM; kc += 64) {
          __builtin_prefetch(Wb + kc + 512, 0, 0);   // global_prefetch_b8
          v16bf a0 = cat16(ld8(Wb + kc), ld8(Wb + kc + 16));
          v16bf b0 = cat16(ld8(vec + kc), ld8(vec + kc + 16));
          v16bf a1 = cat16(ld8(Wb + kc + 32), ld8(Wb + kc + 48));
          v16bf b1 = cat16(ld8(vec + kc + 32), ld8(vec + kc + 48));
          acc0 = __builtin_amdgcn_wmma_f32_16x16x32_bf16(false, a0, false, b0,
                                                         (short)0, acc0, false, false);
          acc1 = __builtin_amdgcn_wmma_f32_16x16x32_bf16(false, a1, false, b1,
                                                         (short)0, acc1, false, false);
        }
        // All 16 D columns are identical; lane0 holds rows 0-7, lane16 rows 8-15
        float* dst = (mat == 0) ? gi[g] : gh[g];
        if (lane == 0)  for (int j = 0; j < 8; ++j) dst[j]     = acc0[j] + acc1[j];
        if (lane == 16) for (int j = 0; j < 8; ++j) dst[8 + j] = acc0[j] + acc1[j];
      }
      __syncthreads();

      if (tid < 16) {                          // gate math on this WG's slice
        int j = tid, hj = wg * 16 + j;
        const float* bi = bih + (size_t)l * G3;
        const float* bh = bhh + (size_t)l * G3;
        float ir = gi[0][j] + bi[hj];
        float iz = gi[1][j] + bi[L_DIM + hj];
        float in_ = gi[2][j] + bi[2 * L_DIM + hj];
        float hr = gh[0][j] + bh[hj];
        float hz = gh[1][j] + bh[L_DIM + hj];
        float hn = gh[2][j] + bh[2 * L_DIM + hj];
        float r = 1.f / (1.f + expf(-(ir + hr)));
        float z = 1.f / (1.f + expf(-(iz + hz)));
        float n = tanhf(in_ + r * hn);
        float h = (1.f - z) * n + z * hold[l][j];
        hold[l][j] = h;
        hNew[l * L_DIM + hj] = f2b(h);
        if (l == DEPTH - 1) H5[(size_t)t * L_DIM + hj] = f2b(h);
      }
      gridBarrier(barCnt, barGen, NWG);
    }
  }
}

// ---------------------------------------------------------------------------
extern "C" void kernel_launch(void* const* d_in, const int* in_sizes, int n_in,
                              void* d_out, int out_size, void* d_ws, size_t ws_size,
                              hipStream_t stream) {
  const float* x      = (const float*)d_in[0];
  const float* lin1_w = (const float*)d_in[1];
  const float* lin1_b = (const float*)d_in[2];
  const float* w_ih   = (const float*)d_in[3];
  const float* w_hh   = (const float*)d_in[4];
  const float* b_ih   = (const float*)d_in[5];
  const float* b_hh   = (const float*)d_in[6];
  const float* lin2_w = (const float*)d_in[7];
  const float* lin2_b = (const float*)d_in[8];
  float* out = (float*)d_out;

  char* ws = (char*)d_ws;
  size_t off = 0;
  auto carve = [&](size_t bytes) {
    void* p = ws + off;
    off = (off + bytes + 255) & ~(size_t)255;
    return p;
  };
  const size_t nW = (size_t)DEPTH * G3 * L_DIM;
  u16* wIh_bf = (u16*)carve(nW * 2);
  u16* wHh_bf = (u16*)carve(nW * 2);
  u16* l1w_bf = (u16*)carve((size_t)L_DIM * IO_DIM * 2);
  u16* l2w_bf = (u16*)carve((size_t)IO_DIM * L_DIM * 2);
  u16* x_bf   = (u16*)carve((size_t)T_STEPS * IO_DIM * 2);
  u16* A_bf   = (u16*)carve((size_t)T_STEPS * L_DIM * 2);
  float* GI0  = (float*)carve((size_t)T_STEPS * G3 * 4);
  u16* H5_bf  = (u16*)carve((size_t)T_STEPS * L_DIM * 2);
  u16* hPing  = (u16*)carve((size_t)DEPTH * L_DIM * 2);
  u16* hPong  = (u16*)carve((size_t)DEPTH * L_DIM * 2);
  u32* bar    = (u32*)carve(256);

  // 1) weight/input conversion to bf16 (stays resident in 192MB L2)
  f32_to_bf16<<<2048, 256, 0, stream>>>(w_ih, wIh_bf, nW);
  f32_to_bf16<<<2048, 256, 0, stream>>>(w_hh, wHh_bf, nW);
  f32_to_bf16<<<64,   256, 0, stream>>>(lin1_w, l1w_bf, (size_t)L_DIM * IO_DIM);
  f32_to_bf16<<<64,   256, 0, stream>>>(lin2_w, l2w_bf, (size_t)IO_DIM * L_DIM);
  f32_to_bf16<<<128,  256, 0, stream>>>(x, x_bf, (size_t)T_STEPS * IO_DIM);
  init_state<<<24, 256, 0, stream>>>(hPing, hPong, bar);

  // 2) lin1 batched over all T:  A[T,L] = x[T,64] * lin1_w^T (+b), store bf16
  gemm_bf16<<<256, 256, 0, stream>>>(x_bf, l1w_bf, lin1_b, nullptr, A_bf,
                                     T_STEPS, L_DIM, IO_DIM);
  // 3) layer-0 ih gates batched over all T: GI0[T,3L] = A * w_ih[0]^T (bias in core)
  gemm_bf16<<<1024, 256, 0, stream>>>(A_bf, wIh_bf, nullptr, GI0, nullptr,
                                      T_STEPS, G3, L_DIM);
  // 4) sequential 6-layer GRU core (persistent, WMMA GEMV + grid barriers)
  gru_persistent<<<NWG, TPB, 0, stream>>>(wIh_bf, wHh_bf, b_ih, b_hh, GI0,
                                          hPing, hPong, H5_bf, bar, bar + 1);
  // 5) lin2 batched over all T: out[T,64] = H5 * lin2_w^T + b
  gemm_bf16<<<128, 256, 0, stream>>>(H5_bf, l2w_bf, lin2_b, out, nullptr,
                                     T_STEPS, IO_DIM, L_DIM);
}
